// EquivariantSelfAttentionTriton_28321014350247
// MI455X (gfx1250) — matrix operand order repaired and checked by reading, not compiled, not run on hardware
//
#include <hip/hip_runtime.h>
#include <hip/hip_bf16.h>

// ---------------------------------------------------------------------------
// EquivariantSelfAttention for MI455X (gfx1250, wave32, WMMA f16 16x16x32)
// B=4, N=2048, HID=256, HEADS=8, HD=32
// ---------------------------------------------------------------------------

typedef _Float16 half_t;
typedef __attribute__((ext_vector_type(16))) _Float16 v16h;
typedef __attribute__((ext_vector_type(8)))  _Float16 v8h;
typedef __attribute__((ext_vector_type(8)))  float    v8f;
typedef __attribute__((ext_vector_type(4)))  int      v4i;

#define BB   4
#define NN   2048
#define HID  256
#define NH   8
#define HD   32
#define TOK  (BB * NN)          // 8192
#define DV   128                // v(32) + 3*vec(96) per head

// ---- CDNA5 async global->LDS path (guarded; falls back to ld+ds_store) ----
#if defined(__has_builtin)
#if __has_builtin(__builtin_amdgcn_global_load_async_to_lds_b128)
#define ASYNC_LDS 1
#endif
#endif
#ifndef ASYNC_LDS
#define ASYNC_LDS 0
#endif

#if ASYNC_LDS
typedef __attribute__((address_space(1))) v4i gas_v4i;   // v4i in global AS
typedef __attribute__((address_space(3))) v4i las_v4i;   // v4i in LDS AS
static __device__ __forceinline__ void copy16_g2l(const half_t* g, half_t* l) {
    // global flat address == AS1 address; LDS offset == low 32 bits of flat.
    __builtin_amdgcn_global_load_async_to_lds_b128(
        (gas_v4i*)(unsigned long long)g,
        (las_v4i*)(unsigned int)(unsigned long long)l, 0, 0);
}
#else
static __device__ __forceinline__ void copy16_g2l(const half_t* g, half_t* l) {
    *(v8h*)l = *(const v8h*)g;
}
#endif

static __device__ __forceinline__ void wait_async_done() {
#if ASYNC_LDS
#if __has_builtin(__builtin_amdgcn_s_wait_asynccnt)
    __builtin_amdgcn_s_wait_asynccnt(0);
#else
    asm volatile("s_wait_asynccnt 0x0" ::: "memory");
#endif
#endif
}

static __device__ __forceinline__ v8f zero8() {
    v8f z;
#pragma unroll
    for (int i = 0; i < 8; i++) z[i] = 0.0f;
    return z;
}

static __device__ __forceinline__ v16h pack16(v8h lo, v8h hi) {
    v16h r;
#pragma unroll
    for (int i = 0; i < 8; i++) { r[i] = lo[i]; r[i + 8] = hi[i]; }
    return r;
}

// ---------------------------------------------------------------------------
// Weight f32 -> f16 convert
// ---------------------------------------------------------------------------
__global__ void cvt_f16_kernel(const float* __restrict__ src,
                               half_t* __restrict__ dst, int n) {
    int i = blockIdx.x * 256 + threadIdx.x;
    if (i < n) dst[i] = (half_t)src[i];
}

// ---------------------------------------------------------------------------
// Prep: split x into scalar/vec halves (f16), vec -> VT (transposed v_all
// layout for attention), per-channel vec L2 norm.
// ---------------------------------------------------------------------------
__global__ void prep_kernel(const float* __restrict__ x,
                            half_t* __restrict__ XsH,
                            half_t* __restrict__ VecH,
                            half_t* __restrict__ VT,
                            float* __restrict__ vnorm) {
    long long idx = (long long)blockIdx.x * 256 + threadIdx.x;  // TOK*HID
    long long t  = idx >> 8;
    int ch       = (int)(idx & 255);
    long long b  = t >> 11;
    long long n  = t & 2047;
    const float* xp = x + t * 4 * HID + ch;   // x[b,n,s,ch]
    float xs = xp[0];
    float v0 = xp[HID], v1 = xp[2 * HID], v2 = xp[3 * HID];
    XsH[t * HID + ch] = (half_t)xs;
    VecH[(t * 3 + 0) * HID + ch] = (half_t)v0;
    VecH[(t * 3 + 1) * HID + ch] = (half_t)v1;
    VecH[(t * 3 + 2) * HID + ch] = (half_t)v2;
    vnorm[t * HID + ch] = sqrtf(v0 * v0 + v1 * v1 + v2 * v2);
    int h = ch >> 5, d = ch & 31;
    long long bh = b * NH + h;
    VT[((bh * DV + 32 + 0 * 32 + d) << 11) + n] = (half_t)v0;
    VT[((bh * DV + 32 + 1 * 32 + d) << 11) + n] = (half_t)v1;
    VT[((bh * DV + 32 + 2 * 32 + d) << 11) + n] = (half_t)v2;
}

// ---------------------------------------------------------------------------
// Generic 16x16-tile WMMA GEMM:  C = A(MxK) * W(NoutxK)^T + bias
// MODE 0: f16 out, *scale   MODE 1: scatter into VT (v_all transposed)
// MODE 2: f32 sigmoid out   MODE 3: f32 out
// ---------------------------------------------------------------------------
template <int KDIM, int MODE>
__global__ __launch_bounds__(256) void gemm16(
    const half_t* __restrict__ A, const half_t* __restrict__ W,
    const float* __restrict__ bias, void* __restrict__ C,
    int M, int Nout, float scale) {
    const int lane = threadIdx.x & 31;
    const int wave = threadIdx.x >> 5;
    const int ntn  = Nout >> 4;
    long long tile = (long long)blockIdx.x * 8 + wave;
    if (tile >= (long long)(M >> 4) * ntn) return;
    const int tm = (int)(tile / ntn);
    const int tn = (int)(tile % ntn);
    const int r  = lane & 15;     // A: row   B: column
    const int g  = lane >> 4;     // lane-group K offset
    const half_t* ap = A + (long long)(tm * 16 + r) * KDIM + g * 16;
    const half_t* bp = W + (long long)(tn * 16 + r) * KDIM + g * 16;
    v8f acc = zero8();
#pragma unroll 4
    for (int k = 0; k < KDIM; k += 32) {
        v8h  alo = *(const v8h*)(ap + k);
        v8h  ahi = *(const v8h*)(ap + k + 8);
        v16h av  = pack16(alo, ahi);
        v16h bv  = *(const v16h*)(bp + k);
        acc = __builtin_amdgcn_wmma_f32_16x16x32_f16(
            false, av, false, bv, (short)0, acc, false, false);
    }
    const int   ocol = tn * 16 + r;
    const float bb   = bias ? bias[ocol] : 0.0f;
#pragma unroll
    for (int i = 0; i < 8; i++) {
        const long long orow = (long long)tm * 16 + g * 8 + i;
        float v = acc[i] + bb;
        if (MODE == 0) {
            ((half_t*)C)[orow * Nout + ocol] = (half_t)(v * scale);
        } else if (MODE == 1) {  // V projection -> VT[bh][d][n]
            long long b = orow >> 11, n = orow & 2047;
            int h = ocol >> 5, d = ocol & 31;
            ((half_t*)C)[(((b * NH + h) * DV + d) << 11) + n] = (half_t)v;
        } else if (MODE == 2) {
            ((float*)C)[orow * Nout + ocol] = 1.0f / (1.0f + __expf(-v));
        } else {
            ((float*)C)[orow * Nout + ocol] = v;
        }
    }
}

// ---------------------------------------------------------------------------
// Invariants: vec_dot from vec_proj halves, build f16 invariant row (alpha'd)
// ---------------------------------------------------------------------------
__global__ void invariants_kernel(const half_t* __restrict__ VP,
                                  const float* __restrict__ vnorm,
                                  const float* __restrict__ alpha_dot,
                                  const float* __restrict__ alpha_norm,
                                  float* __restrict__ vdot,
                                  half_t* __restrict__ Inv) {
    long long idx = (long long)blockIdx.x * 256 + threadIdx.x;
    long long t = idx >> 8;
    int ch      = (int)(idx & 255);
    float s = 0.0f;
#pragma unroll
    for (int c = 0; c < 3; c++) {
        long long base = (t * 3 + c) * 512;
        s += (float)VP[base + ch] * (float)VP[base + 256 + ch];
    }
    vdot[t * HID + ch] = s;
    Inv[t * 512 + ch]       = (half_t)(s * alpha_dot[0]);
    Inv[t * 512 + 256 + ch] = (half_t)(vnorm[t * HID + ch] * alpha_norm[0]);
}

// ---------------------------------------------------------------------------
// Flash attention: one block per (b,h,128-q-chunk); each wave owns 16 q rows.
// K (64x32, 4KB) and V (128x64, 16KB) tiles are staged into LDS ONCE per
// workgroup (8x dedup vs per-wave global loads) with async global->LDS copies
// (ASYNCcnt), double-buffered so the DMA of tile i+1 overlaps the WMMAs of
// tile i. Scores: 4 WMMAs from LDS K. P goes through LDS for the C->A layout
// change, then 16 WMMAs against LDS V.
// ---------------------------------------------------------------------------
__global__ __launch_bounds__(256) void attn_kernel(
    const half_t* __restrict__ Qh, const half_t* __restrict__ Kh,
    const half_t* __restrict__ VT, float* __restrict__ OutAll,
    half_t* __restrict__ XoutH) {
    __shared__ __align__(16) half_t Ktile[2][64][32];   // 2 x 4KB
    __shared__ __align__(16) half_t Vtile[2][128][64];  // 2 x 16KB
    __shared__ __align__(16) half_t P[8][16][72];       // per-wave P, padded

    const int lane = threadIdx.x & 31;
    const int wave = threadIdx.x >> 5;
    const int tid  = threadIdx.x;
    const int bid  = blockIdx.x;          // (b*NH+h)*16 + qchunk
    const int qch  = bid & 15;
    const int bh   = bid >> 4;
    const int b    = bh >> 3;
    const int h    = bh & 7;
    const int q0   = qch * 128 + wave * 16;
    const int r    = lane & 15;
    const int g    = lane >> 4;
    const long long bN = (long long)b * NN;

    // resident Q A-tile (rows q0..q0+15, d = 0..31)
    const half_t* qp = Qh + (bN + q0 + r) * HID + h * HD + g * 16;
    v16h qa = pack16(*(const v8h*)qp, *(const v8h*)(qp + 8));

    const half_t* kbase = Kh + bN * HID + h * HD;           // + key*HID + d
    const half_t* vbase = VT + ((long long)bh * DV) * NN;   // + d*NN + key

    // cooperative stage of one 64-key K tile + V tile into LDS buffer sb
    auto stage = [&](int sb, int sk) {
        {   // K: 64 rows x 32 halves; 256 threads x 16B
            int kr = tid >> 2, kc = (tid & 3) * 8;
            copy16_g2l(kbase + (long long)(sk + kr) * HID + kc,
                       &Ktile[sb][kr][kc]);
        }
#pragma unroll
        for (int c = 0; c < 4; c++) {  // V: 128 rows x 64 halves; 4 x 16B/thr
            int i2 = tid * 4 + c;
            int vr = i2 >> 3, vc = (i2 & 7) * 8;
            copy16_g2l(vbase + (long long)vr * NN + sk + vc,
                       &Vtile[sb][vr][vc]);
        }
    };

    float m[8], l[8];
#pragma unroll
    for (int i = 0; i < 8; i++) { m[i] = -3.0e38f; l[i] = 0.0f; }
    v8f O[8];
#pragma unroll
    for (int j = 0; j < 8; j++) O[j] = zero8();

    stage(0, 0);
    wait_async_done();
    __syncthreads();

    int buf = 0;
    for (int k0 = 0; k0 < NN; k0 += 64) {
        if (k0 + 64 < NN) stage(buf ^ 1, k0 + 64);  // overlap next tile's DMA

        // ---- scores: 4 x (16q x 16k) tiles from LDS K, K-dim = d = 32
        v8f S[4];
#pragma unroll
        for (int t = 0; t < 4; t++) {
            v16h kv = *(const v16h*)&Ktile[buf][t * 16 + r][g * 16];
            S[t] = __builtin_amdgcn_wmma_f32_16x16x32_f16(
                false, qa, false, kv, (short)0, zero8(), false, false);
        }

        // ---- online softmax: row stats across the 16 lanes holding a row
        float rs[8];
#pragma unroll
        for (int i = 0; i < 8; i++) {
            float v = fmaxf(fmaxf(S[0][i], S[1][i]), fmaxf(S[2][i], S[3][i]));
#pragma unroll
            for (int off = 1; off < 16; off <<= 1)
                v = fmaxf(v, __shfl_xor(v, off, 16));
            float mn = fmaxf(m[i], v);
            float a  = __expf(m[i] - mn);
            m[i] = mn;
            l[i] *= a;
#pragma unroll
            for (int j = 0; j < 8; j++) O[j][i] *= a;
            rs[i] = 0.0f;
        }
#pragma unroll
        for (int t = 0; t < 4; t++)
#pragma unroll
            for (int i = 0; i < 8; i++) {
                float p = __expf(S[t][i] - m[i]);
                rs[i] += p;
                P[wave][g * 8 + i][t * 16 + r] = (half_t)p;  // row-major P
            }
#pragma unroll
        for (int i = 0; i < 8; i++) {
            float v = rs[i];
#pragma unroll
            for (int off = 1; off < 16; off <<= 1) v += __shfl_xor(v, off, 16);
            l[i] += v;
        }
        __syncthreads();  // P C-layout stores -> A-layout loads (cross-lane)

        // ---- repack P as WMMA A operands (two K=32 slabs of the 64 keys)
        v16h Pa[2];
#pragma unroll
        for (int t2 = 0; t2 < 2; t2++) {
            v8h plo = *(const v8h*)&P[wave][r][t2 * 32 + g * 16];
            v8h phi = *(const v8h*)&P[wave][r][t2 * 32 + g * 16 + 8];
            Pa[t2] = pack16(plo, phi);
        }

        // ---- O += P @ V   (8 d-tiles x 2 k-slabs) from LDS V
#pragma unroll
        for (int j = 0; j < 8; j++) {
#pragma unroll
            for (int t2 = 0; t2 < 2; t2++) {
                v16h vv = *(const v16h*)&Vtile[buf][j * 16 + r][t2 * 32 + g * 16];
                O[j] = __builtin_amdgcn_wmma_f32_16x16x32_f16(
                    false, Pa[t2], false, vv, (short)0, O[j], false, false);
            }
        }

        wait_async_done();  // next buffer fully staged
        __syncthreads();    // all reads of buf/P done before reuse
        buf ^= 1;
    }

    // ---- normalize + store out_all (f32) and x_out (f16, d<32 section)
#pragma unroll
    for (int i = 0; i < 8; i++) {
        float inv = 1.0f / l[i];
        int qr = q0 + g * 8 + i;
#pragma unroll
        for (int j = 0; j < 8; j++) {
            float v = O[j][i] * inv;
            int d = j * 16 + r;
            OutAll[((long long)bh * NN + qr) * DV + d] = v;
            if (j < 2)
                XoutH[(bN + qr) * HID + h * HD + d] = (half_t)v;
        }
    }
}

// ---------------------------------------------------------------------------
// Final combine: x_updated + gated vec channels -> d_out (B,N,4,HID) f32
// ---------------------------------------------------------------------------
__global__ void combine_kernel(const float* __restrict__ o,
                               const float* __restrict__ vdot,
                               const float* __restrict__ vnorm,
                               const float* __restrict__ gate,
                               const float* __restrict__ OutAll,
                               const float* __restrict__ x,
                               float* __restrict__ out) {
    long long idx = (long long)blockIdx.x * 256 + threadIdx.x;
    long long t = idx >> 8;
    int ch      = (int)(idx & 255);
    long long b = t >> 11, n = t & 2047;
    float xu = vdot[t * HID + ch] * o[t * 768 + ch] +
               vnorm[t * HID + ch] * o[t * 768 + 256 + ch] +
               o[t * 768 + 512 + ch];
    out[(t * 4 + 0) * HID + ch] = xu;
    int h = ch >> 5, d = ch & 31;
    long long bh = b * NH + h;
    float gt = gate[t * HID + ch];
#pragma unroll
    for (int c = 0; c < 3; c++) {
        float aggr = OutAll[(bh * NN + n) * DV + 32 + c * 32 + d];
        out[(t * 4 + 1 + c) * HID + ch] =
            gt * aggr + x[(t * 4 + 1 + c) * HID + ch];
    }
}

// ---------------------------------------------------------------------------
// Launch
// ---------------------------------------------------------------------------
extern "C" void kernel_launch(void* const* d_in, const int* in_sizes, int n_in,
                              void* d_out, int out_size, void* d_ws,
                              size_t ws_size, hipStream_t stream) {
    const float* x     = (const float*)d_in[0];
    const float* Wq    = (const float*)d_in[1];
    const float* bq    = (const float*)d_in[2];
    const float* Wk    = (const float*)d_in[3];
    const float* bk    = (const float*)d_in[4];
    const float* Wv    = (const float*)d_in[5];
    const float* bv    = (const float*)d_in[6];
    const float* Wvec  = (const float*)d_in[7];
    const float* Wo    = (const float*)d_in[8];
    const float* bo    = (const float*)d_in[9];
    const float* Wg    = (const float*)d_in[10];
    const float* bg    = (const float*)d_in[11];
    const float* a_dot = (const float*)d_in[12];
    const float* a_nrm = (const float*)d_in[13];
    float* out = (float*)d_out;

    // --- workspace carve-out
    char* cur = (char*)d_ws;
    auto alloc = [&](size_t bytes) -> void* {
        void* p = cur;
        cur += (bytes + 255) & ~(size_t)255;
        return p;
    };
    half_t* XsH   = (half_t*)alloc((size_t)TOK * HID * 2);
    half_t* VecH  = (half_t*)alloc((size_t)TOK * 3 * HID * 2);
    half_t* QH    = (half_t*)alloc((size_t)TOK * HID * 2);
    half_t* KH    = (half_t*)alloc((size_t)TOK * HID * 2);
    half_t* VT    = (half_t*)alloc((size_t)BB * NH * DV * NN * 2);
    half_t* VP    = (half_t*)alloc((size_t)TOK * 3 * 512 * 2);
    float*  vnorm = (float*)alloc((size_t)TOK * HID * 4);
    float*  vdot  = (float*)alloc((size_t)TOK * HID * 4);
    half_t* Inv   = (half_t*)alloc((size_t)TOK * 512 * 2);
    float*  gate  = (float*)alloc((size_t)TOK * HID * 4);
    float*  OutA  = (float*)alloc((size_t)BB * NH * NN * DV * 4);
    half_t* XoutH = (half_t*)alloc((size_t)TOK * HID * 2);
    float*  oBuf  = (float*)alloc((size_t)TOK * 768 * 4);
    half_t* WqH   = (half_t*)alloc(65536 * 2);
    half_t* WkH   = (half_t*)alloc(65536 * 2);
    half_t* WvH   = (half_t*)alloc(65536 * 2);
    half_t* WvecH = (half_t*)alloc(131072 * 2);
    half_t* WgH   = (half_t*)alloc(131072 * 2);
    half_t* WoH   = (half_t*)alloc(196608 * 2);

    // --- weight conversion
    cvt_f16_kernel<<<256, 256, 0, stream>>>(Wq, WqH, 65536);
    cvt_f16_kernel<<<256, 256, 0, stream>>>(Wk, WkH, 65536);
    cvt_f16_kernel<<<256, 256, 0, stream>>>(Wv, WvH, 65536);
    cvt_f16_kernel<<<512, 256, 0, stream>>>(Wvec, WvecH, 131072);
    cvt_f16_kernel<<<512, 256, 0, stream>>>(Wg, WgH, 131072);
    cvt_f16_kernel<<<768, 256, 0, stream>>>(Wo, WoH, 196608);

    // --- prep
    prep_kernel<<<TOK, 256, 0, stream>>>(x, XsH, VecH, VT, vnorm);

    const float qscale = 0.17677669529663687f;  // 1/sqrt(32)

    // --- projections (Q scaled; V scattered into transposed v_all)
    gemm16<256, 0><<<1024, 256, 0, stream>>>(XsH, WqH, bq, QH, TOK, 256, qscale);
    gemm16<256, 0><<<1024, 256, 0, stream>>>(XsH, WkH, bk, KH, TOK, 256, 1.0f);
    gemm16<256, 1><<<1024, 256, 0, stream>>>(XsH, WvH, bv, VT, TOK, 256, 1.0f);
    gemm16<256, 0><<<6144, 256, 0, stream>>>(VecH, WvecH, nullptr, VP,
                                             TOK * 3, 512, 1.0f);

    // --- invariants + gate
    invariants_kernel<<<TOK, 256, 0, stream>>>(VP, vnorm, a_dot, a_nrm, vdot, Inv);
    gemm16<512, 2><<<1024, 256, 0, stream>>>(Inv, WgH, bg, gate, TOK, 256, 1.0f);

    // --- flash attention over concat'd values
    attn_kernel<<<BB * NH * (NN / 128), 256, 0, stream>>>(QH, KH, VT, OutA, XoutH);

    // --- output projection + final combine
    gemm16<256, 3><<<3072, 256, 0, stream>>>(XoutH, WoH, bo, oBuf, TOK, 768, 1.0f);
    combine_kernel<<<TOK, 256, 0, stream>>>(oBuf, vdot, vnorm, gate, OutA, x, out);
}
